// LocalSpatialAttention_70231305225165
// MI455X (gfx1250) — matrix operand order repaired and checked by reading, not compiled, not run on hardware
//
#include <hip/hip_runtime.h>
#include <hip/hip_bf16.h>
#include <math.h>

// ---------------------------------------------------------------------------
// Types for CDNA5 WMMA (wave32): v_wmma_f32_16x16x32_bf16
// ---------------------------------------------------------------------------
typedef __attribute__((ext_vector_type(16))) __bf16 v16bf;
typedef __attribute__((ext_vector_type(8)))  float  v8f;
typedef __attribute__((ext_vector_type(4)))  unsigned int v4u;
typedef __attribute__((ext_vector_type(8)))  int    v8i;
typedef __attribute__((ext_vector_type(4)))  int    v4i;

#define HW   4096
#define CHN  256
#define CQK  64
#define KDIM 2304   // 256*9 im2col depth
#define NKSTEP (KDIM / 32)   // 72
#define NJ   (HW / 32)       // 128

// --- gfx1250 feature probes (fall back to plain global loads if absent) ----
#if defined(__AMDGCN__) && __has_builtin(__builtin_amdgcn_tensor_load_to_lds) && \
    __has_builtin(__builtin_amdgcn_s_wait_tensorcnt)
#define USE_TDM 1
#else
#define USE_TDM 0
#endif

#if defined(__AMDGCN__) && __has_builtin(__builtin_amdgcn_global_load_async_to_lds_b128) && \
    __has_builtin(__builtin_amdgcn_s_wait_asynccnt)
#define USE_ASYNC 1
#else
#define USE_ASYNC 0
#endif

__device__ __forceinline__ v8f wmma_bf16(v16bf a, v16bf b, v8f c) {
  return __builtin_amdgcn_wmma_f32_16x16x32_bf16(
      /*neg_a=*/false, a, /*neg_b=*/false, b,
      /*c_mod=*/(short)0, c, /*reuse_a=*/false, /*reuse_b=*/false);
}

// A-matrix 16x32 bf16 fragment from row-major [M][K] memory.
// ISA layout: lanes 0-15 -> M=lane, K pairs {0..7, 16..23};
//             lanes 16-31 -> M=lane-16, K pairs {8..15, 24..31}.
__device__ __forceinline__ v16bf load_a_frag(const __bf16* base, int stride) {
  const int lane = threadIdx.x & 31;
  const int r    = lane & 15;
  const int hi   = (lane >> 4) & 1;
  const __bf16* p = base + (size_t)r * stride + hi * 8;
  v16bf f;
#pragma unroll
  for (int i = 0; i < 8; ++i) f[i] = p[i];
#pragma unroll
  for (int i = 0; i < 8; ++i) f[i + 8] = p[i + 16];
  return f;
}

// B-matrix 32x16 bf16 fragment from row-major [N][K] memory.
// ISA layout: lane -> column N; lanes 0-15: K {0..7, 16..23}, lanes 16-31: K {8..15, 24..31}.
__device__ __forceinline__ v16bf load_b_frag(const __bf16* base, int stride) {
  const int lane = threadIdx.x & 31;
  const int r    = lane & 15;
  const int hi   = (lane >> 4) & 1;
  const __bf16* p = base + (size_t)r * stride + hi * 8;
  v16bf f;
#pragma unroll
  for (int i = 0; i < 8; ++i) f[i] = p[i];
#pragma unroll
  for (int i = 0; i < 8; ++i) f[i + 8] = p[i + 16];
  return f;
}

// Half-wave (16-lane) reductions for softmax rows (C/D rows live in 16 lanes).
__device__ __forceinline__ float hmax16(float v) {
#pragma unroll
  for (int m = 1; m < 16; m <<= 1) v = fmaxf(v, __shfl_xor(v, m, 32));
  return v;
}
__device__ __forceinline__ float hsum16(float v) {
#pragma unroll
  for (int m = 1; m < 16; m <<= 1) v += __shfl_xor(v, m, 32);
  return v;
}

#if USE_TDM
// Issue a TDM DMA of W[0:384, k0:k0+32] (row stride KDIM) into LDS (packed
// [384][32] bf16 rows). Descriptor per CDNA5 ISA 8.3/8.4.
// amdgpu-toolchain form: 6 args (g0, g1, g2, g3, g4, cpol).
__device__ __forceinline__ void tdm_issue_w(const __bf16* Wg, int k0, __bf16* lds_dst) {
  unsigned long long ga = (unsigned long long)(uintptr_t)(Wg + k0);
  unsigned int la = (unsigned int)(uintptr_t)lds_dst; // low 32b = LDS byte offset
  v4u g0;
  g0[0] = 1u;                                        // count=1, user descriptor
  g0[1] = la;                                        // lds_addr
  g0[2] = (unsigned int)ga;                          // global_addr[31:0]
  g0[3] = (unsigned int)(ga >> 32) | 0x80000000u;    // global_addr[56:32] | type=2
  v8i g1;
  g1[0] = 0x00010000;          // workgroup_mask=0, data_size=1 (2 bytes)
  g1[1] = 0x09000000;          // tensor_dim0[15:0]=2304 in bits 63:48
  g1[2] = 0x01800000;          // tensor_dim1[15:0]=384 in bits 95:80
  g1[3] = 0x00200000;          // tile_dim0=32 in bits 127:112
  g1[4] = 0x00000180;          // tile_dim1=384
  g1[5] = 0x00000900;          // tensor_dim0_stride=2304
  g1[6] = 0; g1[7] = 0;
  v4i gz4 = {0, 0, 0, 0};
  v8i gz8 = {0, 0, 0, 0, 0, 0, 0, 0};
  __builtin_amdgcn_tensor_load_to_lds(g0, g1, gz4, gz4, gz8, 0);
}
#endif

#if USE_ASYNC
typedef __attribute__((address_space(1))) v4i as1_v4i;
typedef __attribute__((address_space(3))) v4i as3_v4i;
// Copy one 32x64-bf16 K tile (4 KB, contiguous) global -> LDS: 8 x b128/lane.
__device__ __forceinline__ void async_copy_k_tile(const __bf16* gsrc, __bf16* ldst) {
  const int lane = threadIdx.x & 31;
#pragma unroll
  for (int i = 0; i < 8; ++i) {
    int byteoff = (i * 32 + lane) * 16;
    __builtin_amdgcn_global_load_async_to_lds_b128(
        (as1_v4i*)((char*)gsrc + byteoff),
        (as3_v4i*)((char*)ldst + byteoff), 0, 0);
  }
}
#endif

// ---------------------------------------------------------------------------
// Kernel 1: pack fp32 conv weights into one bf16 GEMM matrix W[384][2304]
// ---------------------------------------------------------------------------
__global__ void pack_kernel(const float* __restrict__ qw, const float* __restrict__ qb,
                            const float* __restrict__ kw, const float* __restrict__ kb,
                            const float* __restrict__ vw, const float* __restrict__ vb,
                            __bf16* __restrict__ W, float* __restrict__ bias)
{
  int idx = blockIdx.x * 256 + threadIdx.x;
  const int NW = 384 * KDIM;
  if (idx < NW) {
    int m = idx / KDIM;
    int k = idx - m * KDIM;
    float w;
    if (m < 64)       w = qw[(size_t)m * KDIM + k];
    else if (m < 128) w = kw[(size_t)(m - 64) * KDIM + k];
    else              w = vw[(size_t)(m - 128) * KDIM + k];
    W[idx] = (__bf16)w;
  }
  if (idx < 384) {
    bias[idx] = (idx < 64) ? qb[idx] : (idx < 128) ? kb[idx - 64] : vb[idx - 128];
  }
}

// ---------------------------------------------------------------------------
// Kernel 2: fused QKV 3x3 conv as implicit GEMM via WMMA.
// W k-slab [384][32] DMA'd to LDS by the Tensor Data Mover (double-buffered,
// transfer of step n+1 overlaps compute of step n).
// ---------------------------------------------------------------------------
__global__ __launch_bounds__(256) void qkv_conv_kernel(
    const float* __restrict__ xin, const __bf16* __restrict__ W,
    const float* __restrict__ bias,
    __bf16* __restrict__ Qb, __bf16* __restrict__ Kb, __bf16* __restrict__ Vb)
{
  __shared__ __bf16 Bt[16 * 32];           // im2col tile: 16 pixels x 32 k
#if USE_TDM
  __shared__ __bf16 Wl[2][384 * 32];       // double-buffered weight slab
#endif

  const int lane = threadIdx.x & 31;
  const int wave = threadIdx.x >> 5;
  const int tid  = threadIdx.x;
  const int b    = blockIdx.y;
  const int i0   = blockIdx.x * 16;

  v8f acc[3] = {{}, {}, {}};

#if USE_TDM
  if (wave == 0) {
    tdm_issue_w(W, 0, &Wl[0][0]);
    tdm_issue_w(W, 32, &Wl[1][0]);
  }
#endif

  for (int step = 0; step < NKSTEP; ++step) {
    const int k0 = step * 32;
#if USE_TDM
    if (wave == 0) {                       // buffer `step` ready before barrier
      if (step + 1 < NKSTEP) __builtin_amdgcn_s_wait_tensorcnt((short)1);
      else                   __builtin_amdgcn_s_wait_tensorcnt((short)0);
    }
#endif
    // Cooperative im2col: 512 bf16 elements, 2 per thread, zero-padded edges.
    for (int e = tid; e < 512; e += 256) {
      int p  = e >> 5, kk = e & 31;
      int k  = k0 + kk;
      int ch = k / 9;
      int t  = k - ch * 9;
      int ky = t / 3, kx = t - ky * 3;
      int pix = i0 + p;
      int py = pix >> 6, px = pix & 63;
      int y = py + ky - 1, x = px + kx - 1;
      float val = 0.0f;
      if ((unsigned)y < 64u && (unsigned)x < 64u)
        val = xin[(((size_t)b * CHN + ch) * 64 + y) * 64 + x];
      Bt[p * 32 + kk] = (__bf16)val;
    }
    __syncthreads();

    v16bf bfrag = load_b_frag(Bt, 32);
#pragma unroll
    for (int t = 0; t < 3; ++t) {
      int mbase = (wave + 8 * t) * 16;
#if USE_TDM
      v16bf afrag = load_a_frag(&Wl[step & 1][0] + mbase * 32, 32);
#else
      v16bf afrag = load_a_frag(W + (size_t)mbase * KDIM + k0, KDIM);
#endif
      acc[t] = wmma_bf16(afrag, bfrag, acc[t]);
    }
    __syncthreads();
#if USE_TDM
    if (wave == 0 && step + 2 < NKSTEP)    // buffer free: start DMA of step+2
      tdm_issue_w(W, k0 + 64, &Wl[step & 1][0]);
#endif
  }

  // Epilogue: bias + route to Q / K / V buffers (bf16).
  const int colp = lane & 15;
  const int rofs = (lane < 16) ? 0 : 8;
  const int pix  = i0 + colp;
#pragma unroll
  for (int t = 0; t < 3; ++t) {
    int mbase = (wave + 8 * t) * 16;
#pragma unroll
    for (int v = 0; v < 8; ++v) {
      int m = mbase + v + rofs;
      float y = acc[t][v] + bias[m];
      __bf16 yb = (__bf16)y;
      if (m < 64)
        Qb[((size_t)b * HW + pix) * CQK + m] = yb;
      else if (m < 128)
        Kb[((size_t)b * HW + pix) * CQK + (m - 64)] = yb;
      else
        Vb[((size_t)b * CHN + (m - 128)) * HW + pix] = yb;
    }
  }
}

// ---------------------------------------------------------------------------
// Kernel 3: flash attention. Workgroup = (batch, 16-query tile).
// Wave 0: S = Q^T K with K tiles streamed to LDS by async-to-LDS loads
// (double-buffered, ASYNCcnt-synchronized); online softmax; P -> LDS.
// All waves: O[c-tile] = alpha*O + V(16x32) x P^T(32x16) via wmma.
// ---------------------------------------------------------------------------
__global__ __launch_bounds__(256) void attn_kernel(
    const __bf16* __restrict__ Qb, const __bf16* __restrict__ Kb,
    const __bf16* __restrict__ Vb, float* __restrict__ out)
{
  __shared__ __bf16 Pl[16 * 32];   // probabilities tile [i][j] bf16
  __shared__ float alpha_s[16];
  __shared__ float l_s[16];
#if USE_ASYNC
  __shared__ __bf16 Kl[2][32 * CQK];   // double-buffered K tile (4 KB each)
#endif

  const int lane = threadIdx.x & 31;
  const int wave = threadIdx.x >> 5;
  const int b    = blockIdx.y;
  const int i0   = blockIdx.x * 16;

  const int c0 = wave * 32;                      // channels c0..c0+31
  const __bf16* Vrow0 = Vb + (size_t)(b * CHN + c0) * HW;
  const __bf16* Vrow1 = Vb + (size_t)(b * CHN + c0 + 16) * HW;
  const __bf16* Kbase = Kb + (size_t)b * HW * CQK;

  v8f o0 = {}; v8f o1 = {};

  v16bf aq0 = {}; v16bf aq1 = {};
  float mrun[8], lrun[8];
#pragma unroll
  for (int v = 0; v < 8; ++v) { mrun[v] = -INFINITY; lrun[v] = 0.0f; }
  if (wave == 0) {
    const __bf16* qbase = Qb + (size_t)(b * HW + i0) * CQK;
    aq0 = load_a_frag(qbase, CQK);        // d = 0..31
    aq1 = load_a_frag(qbase + 32, CQK);   // d = 32..63
#if USE_ASYNC
    async_copy_k_tile(Kbase, &Kl[0][0]);
    async_copy_k_tile(Kbase + 32 * CQK, &Kl[1][0]);
#endif
  }

  const int colp = lane & 15;
  const int rofs = (lane < 16) ? 0 : 8;

  for (int js = 0; js < NJ; ++js) {
    const int j0 = js * 32;
    if (wave == 0) {
      v8f s0 = {}; v8f s1 = {};
#if USE_ASYNC
      if (js + 1 < NJ) __builtin_amdgcn_s_wait_asynccnt((short)8);  // oldest tile done
      else             __builtin_amdgcn_s_wait_asynccnt((short)0);
      const __bf16* kt = &Kl[js & 1][0];
      s0 = wmma_bf16(aq0, load_b_frag(kt,                 CQK), s0);
      s0 = wmma_bf16(aq1, load_b_frag(kt + 32,            CQK), s0);
      s1 = wmma_bf16(aq0, load_b_frag(kt + 16 * CQK,      CQK), s1);
      s1 = wmma_bf16(aq1, load_b_frag(kt + 16 * CQK + 32, CQK), s1);
#else
      const __bf16* kb0 = Kbase + (size_t)j0 * CQK;
      const __bf16* kb1 = Kbase + (size_t)(j0 + 16) * CQK;
      s0 = wmma_bf16(aq0, load_b_frag(kb0,      CQK), s0);
      s0 = wmma_bf16(aq1, load_b_frag(kb0 + 32, CQK), s0);
      s1 = wmma_bf16(aq0, load_b_frag(kb1,      CQK), s1);
      s1 = wmma_bf16(aq1, load_b_frag(kb1 + 32, CQK), s1);
#endif
#pragma unroll
      for (int v = 0; v < 8; ++v) {
        float sa = s0[v] * 0.125f;               // scale = CQK^-0.5
        float sb = s1[v] * 0.125f;
        float tilemax = hmax16(fmaxf(sa, sb));
        float mnew  = fmaxf(mrun[v], tilemax);
        float al    = __expf(mrun[v] - mnew);
        float p0    = __expf(sa - mnew);
        float p1    = __expf(sb - mnew);
        float rsum  = hsum16(p0 + p1);
        lrun[v] = lrun[v] * al + rsum;
        mrun[v] = mnew;
        int row = v + rofs;
        Pl[row * 32 + colp]      = (__bf16)p0;
        Pl[row * 32 + colp + 16] = (__bf16)p1;
        if (colp == 0) alpha_s[row] = al;
      }
#if USE_ASYNC
      if (js + 2 < NJ)   // frags consumed; stream tile js+2 into freed buffer
        async_copy_k_tile(Kbase + (size_t)(js + 2) * 32 * CQK, &Kl[js & 1][0]);
#endif
    }
    __syncthreads();

    // All waves: rescale O and accumulate V x P^T.
    float al = alpha_s[lane & 15];               // per output column i
    v16bf bp  = load_b_frag(Pl, 32);             // B: [N=i][K=j] from LDS
    v16bf av0 = load_a_frag(Vrow0 + j0, HW);     // A: [M=c][K=j] from global
    v16bf av1 = load_a_frag(Vrow1 + j0, HW);
#pragma unroll
    for (int v = 0; v < 8; ++v) { o0[v] *= al; o1[v] *= al; }
    o0 = wmma_bf16(av0, bp, o0);
    o1 = wmma_bf16(av1, bp, o1);

    if (js + 1 < NJ) {                           // stream-prefetch next V rows
      __builtin_prefetch(Vrow0 + j0 + 32, 0, 1);
      __builtin_prefetch(Vrow1 + j0 + 32, 0, 1);
    }
    __syncthreads();
  }

  if (wave == 0) {
#pragma unroll
    for (int v = 0; v < 8; ++v)
      if (colp == 0) l_s[v + rofs] = lrun[v];
  }
  __syncthreads();

  const float inv = 1.0f / l_s[lane & 15];
  const int icol = i0 + colp;
#pragma unroll
  for (int v = 0; v < 8; ++v) {
    int ch = c0 + v + rofs;
    out[(size_t)(b * CHN + ch) * HW + icol]      = o0[v] * inv;
    out[(size_t)(b * CHN + ch + 16) * HW + icol] = o1[v] * inv;
  }
}

// ---------------------------------------------------------------------------
// Host launcher
// ---------------------------------------------------------------------------
extern "C" void kernel_launch(void* const* d_in, const int* in_sizes, int n_in,
                              void* d_out, int out_size, void* d_ws, size_t ws_size,
                              hipStream_t stream) {
  (void)in_sizes; (void)n_in; (void)out_size; (void)ws_size;
  const float* x  = (const float*)d_in[0];
  const float* qw = (const float*)d_in[1];
  const float* qb = (const float*)d_in[2];
  const float* kw = (const float*)d_in[3];
  const float* kb = (const float*)d_in[4];
  const float* vw = (const float*)d_in[5];
  const float* vb = (const float*)d_in[6];
  float* out = (float*)d_out;

  // Workspace layout (bytes):
  //   W    bf16 [384][2304]      : 1,769,472
  //   bias f32  [384]            : 1,536       @ 1,769,472
  //   Q    bf16 [4][4096][64]    : 2,097,152   @ 1,775,616
  //   K    bf16 [4][4096][64]    : 2,097,152   @ 3,872,768
  //   V    bf16 [4][256][4096]   : 8,388,608   @ 5,969,920   (end ~14.36 MB)
  char* ws = (char*)d_ws;
  __bf16* W    = (__bf16*)(ws);
  float*  bias = (float*)(ws + 1769472);
  __bf16* Qb   = (__bf16*)(ws + 1775616);
  __bf16* Kb   = (__bf16*)(ws + 3872768);
  __bf16* Vb   = (__bf16*)(ws + 5969920);

  pack_kernel<<<(384 * KDIM + 255) / 256, 256, 0, stream>>>(qw, qb, kw, kb, vw, vb, W, bias);
  qkv_conv_kernel<<<dim3(HW / 16, 4), 256, 0, stream>>>(x, W, bias, Qb, Kb, Vb);
  attn_kernel<<<dim3(HW / 16, 4), 256, 0, stream>>>(Qb, Kb, Vb, out);
}
